// GroupedQueryAttention_44727789421253
// MI455X (gfx1250) — matrix operand order repaired and checked by reading, not compiled, not run on hardware
//
#include <hip/hip_runtime.h>
#include <hip/hip_bf16.h>
#include <stdint.h>

// ---------------- problem constants ----------------
#define EMBED   2048
#define NHEADS  32
#define NKVH    8
#define HEADD   64
#define NKV     512            // NKVH * HEADD
#define BATCH   2
#define LQ      2048
#define LKV     2048
#define MBLK    32             // query rows per fused-attention block

typedef __bf16 bf16;
typedef __attribute__((ext_vector_type(16))) __bf16 v16bf;
typedef __attribute__((ext_vector_type(8)))  float  v8f;

__device__ __forceinline__ bf16 f2bf(float x) { return (bf16)x; }
__device__ __forceinline__ v8f zero8() { v8f z = {0.f,0.f,0.f,0.f,0.f,0.f,0.f,0.f}; return z; }

union FragU { uint4 u[2]; v16bf v; };

// ---- A fragment (16x32 bf16, MxK): lane r=lane&15 holds row r;
// lanes 0-15: K=kb..kb+7 (VGPR0-3) and kb+16..kb+23 (VGPR4-7), kb=(lane>>4)*8
__device__ __forceinline__ v16bf load_fragA(const bf16* p, int ld) {
  int lane = threadIdx.x & 31;
  const bf16* q = p + (long)(lane & 15) * ld + ((lane >> 4) << 3);
  FragU f;
  f.u[0] = *(const uint4*)(q);
  f.u[1] = *(const uint4*)(q + 16);
  return f.v;
}

// fp32 source (global or LDS) -> convert to bf16 fragment in-register
__device__ __forceinline__ v16bf load_fragA(const float* p, int ld) {
  int lane = threadIdx.x & 31;
  const float* q = p + (long)(lane & 15) * ld + ((lane >> 4) << 3);
  float4 x0 = *(const float4*)(q);
  float4 x1 = *(const float4*)(q + 4);
  float4 x2 = *(const float4*)(q + 16);
  float4 x3 = *(const float4*)(q + 20);
  v16bf v;
  v[0]=f2bf(x0.x); v[1]=f2bf(x0.y); v[2]=f2bf(x0.z); v[3]=f2bf(x0.w);
  v[4]=f2bf(x1.x); v[5]=f2bf(x1.y); v[6]=f2bf(x1.z); v[7]=f2bf(x1.w);
  v[8]=f2bf(x2.x); v[9]=f2bf(x2.y); v[10]=f2bf(x2.z); v[11]=f2bf(x2.w);
  v[12]=f2bf(x3.x); v[13]=f2bf(x3.y); v[14]=f2bf(x3.z); v[15]=f2bf(x3.w);
  return v;
}

// ---- B fragment (32x16 bf16, KxN), B stored transposed (N x K row-major):
// lane col n=lane&15; K=kb..kb+7 and kb+16..kb+23, kb=(lane>>4)*8
__device__ __forceinline__ v16bf load_fragB(const bf16* p, int ld) {
  int lane = threadIdx.x & 31;
  const bf16* q = p + (long)(lane & 15) * ld + ((lane >> 4) << 3);
  FragU f;
  f.u[0] = *(const uint4*)(q);
  f.u[1] = *(const uint4*)(q + 16);
  return f.v;
}

#define WMMA_BF16(a, b, c) \
  __builtin_amdgcn_wmma_f32_16x16x32_bf16(false, (a), false, (b), (short)0, (c), false, false)

// wave computes a 32x32 tile: 2x2 accumulators, K-loop step 32
template <typename AT>
__device__ __forceinline__ void wave_tile_gemm(const AT* A, int lda,
                                               const bf16* Bt, int ldb,
                                               int K, v8f c[2][2]) {
  for (int k = 0; k < K; k += 32) {
    if (k + 32 < K) {
      __builtin_prefetch((const void*)(A + k + 32), 0, 1);
      __builtin_prefetch((const void*)(Bt + k + 32), 0, 1);
    }
    v16bf a0 = load_fragA(A + k, lda);
    v16bf a1 = load_fragA(A + 16L * lda + k, lda);
    v16bf b0 = load_fragB(Bt + k, ldb);
    v16bf b1 = load_fragB(Bt + 16L * ldb + k, ldb);
    c[0][0] = WMMA_BF16(a0, b0, c[0][0]);
    c[0][1] = WMMA_BF16(a0, b1, c[0][1]);
    c[1][0] = WMMA_BF16(a1, b0, c[1][0]);
    c[1][1] = WMMA_BF16(a1, b1, c[1][1]);
  }
}

// ---- C/D 16x16 f32 tile layout: VGPR i -> M = i + 8*(lane>>4), N = lane&15
__device__ __forceinline__ void store_tile_bf16(bf16* C, int ldc, const v8f c) {
  int lane = threadIdx.x & 31;
  int col = lane & 15, r0 = (lane >> 4) << 3;
  #pragma unroll
  for (int i = 0; i < 8; ++i) C[(long)(r0 + i) * ldc + col] = f2bf(c[i]);
}

__device__ __forceinline__ void store_tile_f32_bias(float* C, int ldc, const v8f c,
                                                    const float* bias) {
  int lane = threadIdx.x & 31;
  int col = lane & 15, r0 = (lane >> 4) << 3;
  float bv = bias[col];
  #pragma unroll
  for (int i = 0; i < 8; ++i) C[(long)(r0 + i) * ldc + col] = c[i] + bv;
}

#define ZERO_ACC(c)                                              \
  const v8f vz_ = {0.f, 0.f, 0.f, 0.f, 0.f, 0.f, 0.f, 0.f};     \
  v8f c[2][2] = {{vz_, vz_}, {vz_, vz_}};

// ================= kernels =================

// Wt[n*K + k] = (bf16) W[k*N + n]   (W: K x N row-major)
__global__ void __launch_bounds__(256)
k_transpose_conv(const float* __restrict__ W, bf16* __restrict__ Wt, int K, int N) {
  long idx = (long)blockIdx.x * 256 + threadIdx.x;
  long n = idx / K;
  int k = (int)(idx - n * K);
  Wt[idx] = f2bf(W[(long)k * N + n]);
}

// Vp: (B*LKV) x 512 bf16  ->  Vt: (B, NKVH, HEADD, LKV) bf16
__global__ void __launch_bounds__(256)
k_transpose_v(const bf16* __restrict__ Vp, bf16* __restrict__ Vt) {
  long idx = (long)blockIdx.x * 256 + threadIdx.x;
  long l = idx & (LKV - 1);
  long r = idx >> 11;
  int d = (int)(r & (HEADD - 1)); r >>= 6;
  int kvh = (int)(r & (NKVH - 1));
  int b = (int)(r >> 3);
  Vt[idx] = Vp[(long)(b * LKV + l) * NKV + kvh * HEADD + d];
}

// C(bf16, MxN) = A(f32, MxK) * Bt(bf16, NxK)^T
__global__ void __launch_bounds__(128)
k_proj_gemm(const float* __restrict__ A, const bf16* __restrict__ Bt,
            bf16* __restrict__ C, int lda, int ldb, int ldc, int K) {
  int wave = threadIdx.x >> 5;
  int m0 = blockIdx.x * 64 + (wave >> 1) * 32;
  int n0 = blockIdx.y * 64 + (wave & 1) * 32;
  ZERO_ACC(c);
  wave_tile_gemm(A + (long)m0 * lda, lda, Bt + (long)n0 * ldb, ldb, K, c);
  #pragma unroll
  for (int i = 0; i < 2; ++i)
    #pragma unroll
    for (int j = 0; j < 2; ++j)
      store_tile_bf16(C + (long)(m0 + 16 * i) * ldc + (n0 + 16 * j), ldc, c[i][j]);
}

// ---------------- fused scores -> softmax -> P*V ----------------
// One block = one (b,h) x 32 query rows. 256 threads (8 waves).
// LDS: 32 x 2048 fp32 score slab (256 KB -- needs CDNA5's 320KB WGP LDS).
__global__ void __launch_bounds__(256)
k_attn_fused(const bf16* __restrict__ Qp, const bf16* __restrict__ Kp,
             const bf16* __restrict__ Vt, float* __restrict__ attnW,
             bf16* __restrict__ AO, const unsigned char* __restrict__ mask,
             float scale) {
  extern __shared__ char smem_raw[];
  float* S = (float*)smem_raw;                       // [MBLK][LKV]

  const int g = blockIdx.z;
  const int b = g >> 5, h = g & 31, kvh = h >> 2;
  const int m0 = blockIdx.x * MBLK;
  const int wave = threadIdx.x >> 5;
  const int lane = threadIdx.x & 31;
  const int col = lane & 15, r0 = (lane >> 4) << 3;

  const bf16* Aq = Qp + ((long)b * LQ + m0) * EMBED + h * HEADD;
  const bf16* Bk = Kp + (long)b * LKV * NKV + kvh * HEADD;
  const bf16* Bv = Vt + (long)(b * NKVH + kvh) * HEADD * LKV;
  const unsigned char* mrow = mask + (long)b * LKV;

  // ---- Phase 1: S = scale * Q K^T (+mask) into LDS ----
  // Q fragments: rows {0-15,16-31} x ksteps {0,32}, held in registers.
  v16bf qa[2][2];
  #pragma unroll
  for (int ti = 0; ti < 2; ++ti)
    #pragma unroll
    for (int ks = 0; ks < 2; ++ks)
      qa[ti][ks] = load_fragA(Aq + (long)ti * 16 * EMBED + ks * 32, EMBED);

  for (int it = 0; it < LKV / (8 * 32); ++it) {      // 8 column tiles per wave
    int n0 = wave * 32 + it * 256;
    v8f c[2][2] = {{zero8(), zero8()}, {zero8(), zero8()}};
    #pragma unroll
    for (int ks = 0; ks < 2; ++ks) {
      v16bf b0 = load_fragB(Bk + (long)n0 * NKV + ks * 32, NKV);
      v16bf b1 = load_fragB(Bk + (long)(n0 + 16) * NKV + ks * 32, NKV);
      c[0][0] = WMMA_BF16(qa[0][ks], b0, c[0][0]);
      c[0][1] = WMMA_BF16(qa[0][ks], b1, c[0][1]);
      c[1][0] = WMMA_BF16(qa[1][ks], b0, c[1][0]);
      c[1][1] = WMMA_BF16(qa[1][ks], b1, c[1][1]);
    }
    #pragma unroll
    for (int ti = 0; ti < 2; ++ti)
      #pragma unroll
      for (int tj = 0; tj < 2; ++tj) {
        int cc = n0 + tj * 16 + col;
        bool msk = (mrow[cc] != 0);
        #pragma unroll
        for (int i = 0; i < 8; ++i) {
          float v = c[ti][tj][i] * scale;
          if (msk) v = -__builtin_inff();
          S[(ti * 16 + r0 + i) * LKV + cc] = v;
        }
      }
  }
  __syncthreads();

  // ---- Phase 2: row softmax in LDS + single pass of normalized weights to HBM ----
  for (int r = wave; r < MBLK; r += 8) {
    float* row = S + (long)r * LKV;
    float* grow = attnW + (long)g * LQ * LKV + (long)(m0 + r) * LKV;
    float m = -3.402823e38f;
    for (int i = 0; i < LKV / 32; ++i) m = fmaxf(m, row[lane + i * 32]);
    #pragma unroll
    for (int o = 16; o; o >>= 1) m = fmaxf(m, __shfl_xor(m, o, 32));
    float s = 0.f;
    for (int i = 0; i < LKV / 32; ++i) {
      int idx = lane + i * 32;
      float e = __expf(row[idx] - m);
      s += e;
      row[idx] = e;
    }
    #pragma unroll
    for (int o = 16; o; o >>= 1) s += __shfl_xor(s, o, 32);
    float inv = 1.f / s;
    for (int i = 0; i < LKV / 32; ++i) {
      int idx = lane + i * 32;
      float v = row[idx] * inv;
      row[idx] = v;        // LDS copy for phase 3
      grow[idx] = v;       // the only HBM pass over attn_weights
    }
  }
  __syncthreads();

  // ---- Phase 3: AO_block(32x64) = P(32x2048, LDS fp32) * V ----
  // K split across the 8 waves (256 keys each); partials reduced via LDS.
  v8f c0[2][2], c1[2][2];
  #pragma unroll
  for (int i = 0; i < 2; ++i)
    #pragma unroll
    for (int j = 0; j < 2; ++j) { c0[i][j] = zero8(); c1[i][j] = zero8(); }

  const int kbase = wave * (LKV / 8);
  for (int k = kbase; k < kbase + LKV / 8; k += 32) {
    v16bf a0 = load_fragA((const float*)S + k, LKV);
    v16bf a1 = load_fragA((const float*)S + 16L * LKV + k, LKV);
    v16bf b0 = load_fragB(Bv + k, LKV);
    v16bf b1 = load_fragB(Bv + 16L * LKV + k, LKV);
    v16bf b2 = load_fragB(Bv + 32L * LKV + k, LKV);
    v16bf b3 = load_fragB(Bv + 48L * LKV + k, LKV);
    c0[0][0] = WMMA_BF16(a0, b0, c0[0][0]);
    c0[0][1] = WMMA_BF16(a0, b1, c0[0][1]);
    c0[1][0] = WMMA_BF16(a1, b0, c0[1][0]);
    c0[1][1] = WMMA_BF16(a1, b1, c0[1][1]);
    c1[0][0] = WMMA_BF16(a0, b2, c1[0][0]);
    c1[0][1] = WMMA_BF16(a0, b3, c1[0][1]);
    c1[1][0] = WMMA_BF16(a1, b2, c1[1][0]);
    c1[1][1] = WMMA_BF16(a1, b3, c1[1][1]);
  }
  __syncthreads();   // done reading P; safe to reuse S for partial sums

  // each wave writes its 32x64 fp32 partial at S[wave*2048 ..]
  float* P = S + wave * (MBLK * HEADD);
  #pragma unroll
  for (int ti = 0; ti < 2; ++ti)
    #pragma unroll
    for (int tj = 0; tj < 2; ++tj)
      #pragma unroll
      for (int i = 0; i < 8; ++i) {
        int rr = ti * 16 + r0 + i;
        P[rr * HEADD + (tj * 16 + col)] = c0[ti][tj][i];
        P[rr * HEADD + (32 + tj * 16 + col)] = c1[ti][tj][i];
      }
  __syncthreads();

  // tree-less reduction over the 8 wave partials, then bf16 store to AO
  bf16* AOb = AO + ((long)b * LQ + m0) * EMBED + h * HEADD;
  for (int e = threadIdx.x; e < MBLK * HEADD; e += 256) {
    float s = 0.f;
    #pragma unroll
    for (int w = 0; w < 8; ++w) s += S[w * (MBLK * HEADD) + e];
    AOb[(long)(e >> 6) * EMBED + (e & 63)] = f2bf(s);
  }
}

// out(f32) = AO(bf16) * WoT^T + bo
__global__ void __launch_bounds__(128)
k_out(const bf16* __restrict__ AO, const bf16* __restrict__ WoT,
      const float* __restrict__ bias, float* __restrict__ out) {
  int wave = threadIdx.x >> 5;
  int m0 = blockIdx.x * 64 + (wave >> 1) * 32;
  int n0 = blockIdx.y * 64 + (wave & 1) * 32;
  ZERO_ACC(c);
  wave_tile_gemm(AO + (long)m0 * EMBED, EMBED, WoT + (long)n0 * EMBED, EMBED, EMBED, c);
  #pragma unroll
  for (int i = 0; i < 2; ++i)
    #pragma unroll
    for (int j = 0; j < 2; ++j)
      store_tile_f32_bias(out + (long)(m0 + 16 * i) * EMBED + (n0 + 16 * j), EMBED,
                          c[i][j], bias + n0 + 16 * j);
}

// ================= launcher =================
extern "C" void kernel_launch(void* const* d_in, const int* in_sizes, int n_in,
                              void* d_out, int out_size, void* d_ws, size_t ws_size,
                              hipStream_t stream) {
  (void)in_sizes; (void)n_in; (void)out_size; (void)ws_size;
  const float* query = (const float*)d_in[0];
  const float* key   = (const float*)d_in[1];
  const float* value = (const float*)d_in[2];
  const unsigned char* mask = (const unsigned char*)d_in[3];
  const float* Wq = (const float*)d_in[4];
  const float* Wk = (const float*)d_in[5];
  const float* Wv = (const float*)d_in[6];
  const float* Wo = (const float*)d_in[7];
  const float* bo = (const float*)d_in[8];

  float* out  = (float*)d_out;
  float* attn = out + (size_t)BATCH * LQ * EMBED;

  uintptr_t ws = (uintptr_t)d_ws;
  auto take = [&](size_t bytes) -> uintptr_t {
    uintptr_t p = ws; ws += (bytes + 255) & ~(size_t)255; return p;
  };
  bf16* WqT = (bf16*)take((size_t)EMBED * EMBED * 2);
  bf16* WkT = (bf16*)take((size_t)NKV * EMBED * 2);
  bf16* WvT = (bf16*)take((size_t)NKV * EMBED * 2);
  bf16* WoT = (bf16*)take((size_t)EMBED * EMBED * 2);
  bf16* Qp  = (bf16*)take((size_t)BATCH * LQ * EMBED * 2);
  bf16* Kp  = (bf16*)take((size_t)BATCH * LKV * NKV * 2);
  bf16* Vp  = (bf16*)take((size_t)BATCH * LKV * NKV * 2);
  bf16* Vt  = (bf16*)take((size_t)BATCH * NKV * LKV * 2);
  bf16* AO  = (bf16*)take((size_t)BATCH * LQ * EMBED * 2);

  // 1. weights: transpose + fp32->bf16
  k_transpose_conv<<<(EMBED * EMBED) / 256, 256, 0, stream>>>(Wq, WqT, EMBED, EMBED);
  k_transpose_conv<<<(EMBED * NKV) / 256, 256, 0, stream>>>(Wk, WkT, EMBED, NKV);
  k_transpose_conv<<<(EMBED * NKV) / 256, 256, 0, stream>>>(Wv, WvT, EMBED, NKV);
  k_transpose_conv<<<(EMBED * EMBED) / 256, 256, 0, stream>>>(Wo, WoT, EMBED, EMBED);

  dim3 blk(128);
  // 2. projections (WMMA)
  k_proj_gemm<<<dim3((BATCH * LQ) / 64, EMBED / 64), blk, 0, stream>>>(
      query, WqT, Qp, EMBED, EMBED, EMBED, EMBED);
  k_proj_gemm<<<dim3((BATCH * LKV) / 64, NKV / 64), blk, 0, stream>>>(
      key, WkT, Kp, EMBED, EMBED, NKV, EMBED);
  k_proj_gemm<<<dim3((BATCH * LKV) / 64, NKV / 64), blk, 0, stream>>>(
      value, WvT, Vp, EMBED, EMBED, NKV, EMBED);

  // 3. V -> (B, kvh, d, Lkv)
  k_transpose_v<<<(BATCH * NKV * LKV) / 256, 256, 0, stream>>>(Vp, Vt);

  // 4. fused scores/softmax/P*V: 256 KB LDS slab per block (CDNA5 320KB WGP LDS)
  constexpr size_t SMEM = (size_t)MBLK * LKV * sizeof(float);   // 262144
  hipFuncSetAttribute((const void*)k_attn_fused,
                      hipFuncAttributeMaxDynamicSharedMemorySize, (int)SMEM);
  k_attn_fused<<<dim3(LQ / MBLK, 1, BATCH * NHEADS), dim3(256), SMEM, stream>>>(
      Qp, Kp, Vt, attn, AO, mask, 0.125f);

  // 5. output projection + bias (WMMA)
  k_out<<<dim3((BATCH * LQ) / 64, EMBED / 64), blk, 0, stream>>>(AO, WoT, bo, out);
}
